// Segmenter_22969485099745
// MI455X (gfx1250) — matrix operand order repaired and checked by reading, not compiled, not run on hardware
//
#include <hip/hip_runtime.h>
#include <stdint.h>

// ---------------------------------------------------------------------------
// Problem constants
// ---------------------------------------------------------------------------
#define NN   8192          // rows of FF / Psi
#define CC   384           // feature dim of FF
#define KK   512           // columns of Psi
#define BSI  8             // images
#define HW   1024          // pixels per image
#define T_SCALE (10.0f / 8192.0f)   // T / N
#define ALPHA 0.05f
#define PW_W  0.1f

// ---------------------------------------------------------------------------
// Workspace layout (bytes, all 256-aligned). Total ~192 MB.
// ---------------------------------------------------------------------------
#define OFF_FFN  ((size_t)0)                 // bf16 FFn            8192*384*2   = 6291456
#define OFF_A    ((size_t)6291456)           // bf16 A              8192*8192*2  = 134217728
#define OFF_DEG  ((size_t)140509184)         // f32 deg             8192*4
#define OFF_DV   ((size_t)140541952)         // f32 D               8192*4
#define OFF_DPT  ((size_t)140574720)         // bf16 DpsiT          512*8192*2   = 8388608
#define OFF_M    ((size_t)148963328)         // f32 M               8192*512*4   = 16777216
#define OFF_R    ((size_t)165740544)         // f32 R               512*512*4    = 1048576
#define OFF_K0   ((size_t)166789120)         // u8 knn cfg0         8*1024*1024
#define OFF_K1   ((size_t)175177728)         // u8 knn cfg1         8*1024*1024
#define OFF_AP   ((size_t)183566336)         // u8 Ap               8*1024*1024

typedef __attribute__((ext_vector_type(16))) __bf16 v16bf;
typedef __attribute__((ext_vector_type(8)))  float  v8f;

__device__ inline uint16_t f2bf(float f) {
    uint32_t u = __float_as_uint(f);
    u += 0x7FFFu + ((u >> 16) & 1u);          // round-to-nearest-even
    return (uint16_t)(u >> 16);
}

// ---------------------------------------------------------------------------
// Small utility kernels
// ---------------------------------------------------------------------------
__global__ void zero_f32_kernel(float* p, int n) {
    int i = blockIdx.x * blockDim.x + threadIdx.x;
    if (i < n) p[i] = 0.0f;
}

// FFn = bf16(FF / ||FF_row||), one block (128 thr) per row
__global__ __launch_bounds__(128) void rownorm_kernel(const float* FF, uint16_t* FFn) {
    __shared__ float red[128];
    int r = blockIdx.x, t = threadIdx.x;
    const float* src = FF + (long)r * CC;
    float s = 0.f;
    for (int c = t; c < CC; c += 128) { float v = src[c]; s += v * v; }
    red[t] = s; __syncthreads();
    for (int o = 64; o > 0; o >>= 1) { if (t < o) red[t] += red[t + o]; __syncthreads(); }
    float inv = 1.0f / sqrtf(red[0]);
    uint16_t* dst = FFn + (long)r * CC;
    for (int c = t; c < CC; c += 128) dst[c] = f2bf(src[c] * inv);
}

// ---------------------------------------------------------------------------
// Pixelwise kNN: one block = 256 rows of one image, feature table in LDS
// ---------------------------------------------------------------------------
template <int KN>
__global__ __launch_bounds__(256) void knn_kernel(const float* im, uint8_t* knn, float dw) {
    __shared__ float sf[5][HW];
    int t = threadIdx.x;
    int b = blockIdx.x >> 2;
    int i = ((blockIdx.x & 3) << 8) + t;
    for (int p = t; p < HW; p += 256) {
        #pragma unroll
        for (int ch = 0; ch < 3; ++ch)
            sf[ch][p] = (im[((long)b * 3 + ch) * HW + p] + 1.0f) * 0.5f;  // PW_DIV = 1
        sf[3][p] = (float)(p & 31) * (dw / 31.0f);   // x = linspace(0,1,32)*dw
        sf[4][p] = (float)(p >> 5) * (dw / 31.0f);   // y
    }
    __syncthreads();
    float f0 = sf[0][i], f1 = sf[1][i], f2 = sf[2][i], f3 = sf[3][i], f4 = sf[4][i];

    float bd[KN]; int bj[KN];
    #pragma unroll
    for (int q = 0; q < KN; ++q) { bd[q] = 3.4e38f; bj[q] = 0; }

    for (int j = 0; j < HW; ++j) {
        if (j == i) continue;                         // diag excluded (-inf in ref)
        float d0 = f0 - sf[0][j], d1 = f1 - sf[1][j], d2 = f2 - sf[2][j];
        float d3 = f3 - sf[3][j], d4 = f4 - sf[4][j];
        float d = d0*d0 + d1*d1 + d2*d2 + d3*d3 + d4*d4;
        if (d < bd[KN - 1]) {                         // stable insertion (ties -> lower idx)
            int pos = KN - 1;
            while (pos > 0 && bd[pos - 1] > d) {
                bd[pos] = bd[pos - 1]; bj[pos] = bj[pos - 1]; --pos;
            }
            bd[pos] = d; bj[pos] = j;
        }
    }
    uint8_t* row = knn + (((long)b) << 20) + ((long)i << 10);
    uint32_t* row32 = (uint32_t*)row;
    for (int w2 = 0; w2 < 256; ++w2) row32[w2] = 0u;
    #pragma unroll
    for (int q = 0; q < KN; ++q) row[bj[q]] = 1;
}

// Ap = (k0 & k0^T) | (k1 & k1^T)
__global__ void mutual_kernel(const uint8_t* k0, const uint8_t* k1, uint8_t* ap) {
    long idx = (long)blockIdx.x * 256 + threadIdx.x;        // < 8*1024*1024
    long b = idx >> 20; long i = (idx >> 10) & 1023; long j = idx & 1023;
    long base = b << 20;
    uint8_t m0 = k0[base + (i << 10) + j] & k0[base + (j << 10) + i];
    uint8_t m1 = k1[base + (i << 10) + j] & k1[base + (j << 10) + i];
    ap[idx] = (uint8_t)(m0 | m1);
}

// ---------------------------------------------------------------------------
// Shared WMMA GEMM helpers: 128x128 block tile, K-step 32, 8 waves (4x2),
// each wave owns a 32x64 sub-tile = 2x4 v_wmma_f32_16x16x32_bf16 tiles.
// LDS tiles are [128 rows][32 cols] bf16, row stride padded to 40.
// A tile is M-major (row-major), B tile is N-major (so B-frag loads are b128).
// Global->LDS copies use the CDNA5 async pipe (GLOBAL_LOAD_ASYNC_TO_LDS_B128,
// tracked with ASYNCcnt) with double-buffered tiles.
// ---------------------------------------------------------------------------
#define LDT 40

// Issue one 128x32 bf16 tile as 512 async b128 copies (2 per thread).
__device__ __attribute__((always_inline)) inline
void async_tile(const uint16_t* gbase, long ld, uint16_t* lds) {
    const int t = threadIdx.x;
    #pragma unroll
    for (int it = 0; it < 2; ++it) {
        int c = t + it * 256;                    // 0..511
        int row = c >> 2, co = (c & 3) << 3;     // 8 bf16 per chunk
        const uint16_t* g = gbase + (long)row * ld + co;
        uint32_t ldsoff = (uint32_t)(size_t)(lds + row * LDT + co);
        asm volatile("global_load_async_to_lds_b128 %0, %1, off"
                     :: "v"(ldsoff), "v"((uint64_t)(size_t)g)
                     : "memory");
    }
}

union FragU { uint4 q[2]; v16bf v; };

// Load A frag (16x32, rows mbase..mbase+15): lane L<16 -> row L, K {0..7,16..23};
// lane L>=16 -> row L-16, K {8..15,24..31}.
__device__ inline void load_afrag(FragU& f, const uint16_t* As, int mbase, int half, int l) {
    const uint16_t* p = As + (mbase + l) * LDT + half * 8;
    f.q[0] = *(const uint4*)(p);
    f.q[1] = *(const uint4*)(p + 16);
}
// Load B frag (32x16, cols nbase..nbase+15) from N-major LDS:
// lane L<16 -> col L, K {0..7,16..23}; lane L>=16 -> col L-16, K {8..15,24..31}.
__device__ inline void load_bfrag(FragU& f, const uint16_t* Bs, int nbase, int half, int l) {
    const uint16_t* p = Bs + (nbase + l) * LDT + half * 8;
    f.q[0] = *(const uint4*)(p);
    f.q[1] = *(const uint4*)(p + 16);
}

// One K=32 MMA step for a wave's 32x64 sub-tile.
__device__ __attribute__((always_inline)) inline
void mma_step(v8f (&acc)[2][4], const uint16_t* Asb, const uint16_t* Bsb,
              int wy, int wx, int half, int l) {
    FragU a[2], b[4];
    #pragma unroll
    for (int m = 0; m < 2; ++m) load_afrag(a[m], Asb, wy * 32 + m * 16, half, l);
    #pragma unroll
    for (int n = 0; n < 4; ++n) load_bfrag(b[n], Bsb, wx * 64 + n * 16, half, l);
    #pragma unroll
    for (int m = 0; m < 2; ++m)
        #pragma unroll
        for (int n = 0; n < 4; ++n)
            acc[m][n] = __builtin_amdgcn_wmma_f32_16x16x32_bf16(
                false, a[m].v, false, b[n].v, (short)0, acc[m][n], false, false);
}

// ---------------------------------------------------------------------------
// GEMM pass 1: S = FFn @ FFn^T (bf16 in, f32 acc); epilogue: zero diag,
// clamp>=0, +0.1*Ap on diagonal blocks, store bf16 A, atomic row sums -> deg.
// ---------------------------------------------------------------------------
__global__ __launch_bounds__(256) void gemm_a_kernel(const uint16_t* FFn,
                                                     const uint8_t* Ap,
                                                     uint16_t* Aout, float* deg) {
    __shared__ uint16_t As[2][128 * LDT];
    __shared__ uint16_t Bs[2][128 * LDT];
    const int i0 = blockIdx.y * 128, j0 = blockIdx.x * 128;
    const int t = threadIdx.x, lane = t & 31, w = t >> 5;
    const int wy = w >> 1, wx = w & 1, half = lane >> 4, l = lane & 15;

    const v8f zf = {0.f,0.f,0.f,0.f,0.f,0.f,0.f,0.f};
    v8f acc[2][4];
    #pragma unroll
    for (int m = 0; m < 2; ++m)
        #pragma unroll
        for (int n = 0; n < 4; ++n) acc[m][n] = zf;

    const int nk = CC / 32;                       // 12
    async_tile(FFn + (long)i0 * CC, CC, As[0]);
    async_tile(FFn + (long)j0 * CC, CC, Bs[0]);
    async_tile(FFn + (long)i0 * CC + 32, CC, As[1]);
    async_tile(FFn + (long)j0 * CC + 32, CC, Bs[1]);

    for (int kt = 0; kt < nk; ++kt) {
        // 4 async ops per tile-pair per thread; tile kt+1 may still be in flight
        if (kt + 1 < nk) asm volatile("s_wait_asynccnt 0x4" ::: "memory");
        else             asm volatile("s_wait_asynccnt 0x0" ::: "memory");
        __syncthreads();
        mma_step(acc, As[kt & 1], Bs[kt & 1], wy, wx, half, l);
        __syncthreads();
        if (kt + 2 < nk) {
            async_tile(FFn + (long)i0 * CC + (kt + 2) * 32, CC, As[kt & 1]);
            async_tile(FFn + (long)j0 * CC + (kt + 2) * 32, CC, Bs[kt & 1]);
        }
    }

    // epilogue
    float prow[2][8];
    #pragma unroll
    for (int m = 0; m < 2; ++m)
        #pragma unroll
        for (int e = 0; e < 8; ++e) prow[m][e] = 0.f;

    #pragma unroll
    for (int m = 0; m < 2; ++m) {
        #pragma unroll
        for (int e = 0; e < 8; ++e) {
            const int i = i0 + wy * 32 + m * 16 + e + (half ? 8 : 0);
            #pragma unroll
            for (int n = 0; n < 4; ++n) {
                const int j = j0 + wx * 64 + n * 16 + l;
                float v = acc[m][n][e];
                if (i == j) v = 0.f; else v = fmaxf(v, 0.f);
                if ((i >> 10) == (j >> 10))
                    v += PW_W * (float)Ap[((long)(i >> 10) << 20) +
                                          ((long)(i & 1023) << 10) + (j & 1023)];
                Aout[(long)i * NN + j] = f2bf(v);
                prow[m][e] += v;
            }
        }
    }
    #pragma unroll
    for (int m = 0; m < 2; ++m) {
        #pragma unroll
        for (int e = 0; e < 8; ++e) {
            float s = prow[m][e];
            s += __shfl_xor(s, 8, 16);
            s += __shfl_xor(s, 4, 16);
            s += __shfl_xor(s, 2, 16);
            s += __shfl_xor(s, 1, 16);
            if (l == 0) {
                const int i = i0 + wy * 32 + m * 16 + e + (half ? 8 : 0);
                atomicAdd(&deg[i], s);
            }
        }
    }
}

// D = rsqrt(deg / (N-1))
__global__ void dcalc_kernel(const float* deg, float* Dv) {
    int i = blockIdx.x * blockDim.x + threadIdx.x;
    if (i < NN) Dv[i] = rsqrtf(deg[i] * (1.0f / (float)(NN - 1)));
}

// DpsiT[n][j] = bf16(D[j] * Psi[j][n] * T/N), stored N-major (ld = 8192)
__global__ void dpsit_kernel(const float* Psi, const float* Dv, uint16_t* DpsiT) {
    long idx = (long)blockIdx.x * 256 + threadIdx.x;   // < 512*8192
    int n = (int)(idx >> 13), j = (int)(idx & (NN - 1));
    DpsiT[idx] = f2bf(Dv[j] * Psi[(long)j * KK + n] * T_SCALE);
}

// ---------------------------------------------------------------------------
// GEMM pass 2: M[i][n] = D[i] * sum_k A_bf16[i][k] * DpsiT[n][k]
// ---------------------------------------------------------------------------
__global__ __launch_bounds__(256) void gemm_m_kernel(const uint16_t* A,
                                                     const uint16_t* DpsiT,
                                                     const float* Dv, float* M) {
    __shared__ uint16_t As[2][128 * LDT];
    __shared__ uint16_t Bs[2][128 * LDT];
    const int i0 = blockIdx.y * 128, n0 = blockIdx.x * 128;
    const int t = threadIdx.x, lane = t & 31, w = t >> 5;
    const int wy = w >> 1, wx = w & 1, half = lane >> 4, l = lane & 15;

    const v8f zf = {0.f,0.f,0.f,0.f,0.f,0.f,0.f,0.f};
    v8f acc[2][4];
    #pragma unroll
    for (int m = 0; m < 2; ++m)
        #pragma unroll
        for (int n = 0; n < 4; ++n) acc[m][n] = zf;

    const int nk = NN / 32;                      // 256
    async_tile(A + (long)i0 * NN, NN, As[0]);
    async_tile(DpsiT + (long)n0 * NN, NN, Bs[0]);
    async_tile(A + (long)i0 * NN + 32, NN, As[1]);
    async_tile(DpsiT + (long)n0 * NN + 32, NN, Bs[1]);

    for (int kt = 0; kt < nk; ++kt) {
        if (kt + 1 < nk) asm volatile("s_wait_asynccnt 0x4" ::: "memory");
        else             asm volatile("s_wait_asynccnt 0x0" ::: "memory");
        __syncthreads();
        mma_step(acc, As[kt & 1], Bs[kt & 1], wy, wx, half, l);
        __syncthreads();
        if (kt + 2 < nk) {
            async_tile(A + (long)i0 * NN + (kt + 2) * 32, NN, As[kt & 1]);
            async_tile(DpsiT + (long)n0 * NN + (kt + 2) * 32, NN, Bs[kt & 1]);
        }
    }

    #pragma unroll
    for (int m = 0; m < 2; ++m) {
        #pragma unroll
        for (int e = 0; e < 8; ++e) {
            const int i = i0 + wy * 32 + m * 16 + e + (half ? 8 : 0);
            const float di = Dv[i];
            #pragma unroll
            for (int n = 0; n < 4; ++n) {
                const int nn2 = n0 + wx * 64 + n * 16 + l;
                M[(long)i * KK + nn2] = di * acc[m][n][e];
            }
        }
    }
}

// ---------------------------------------------------------------------------
// R = (Psi * T/N)^T @ M   (fp32, tiled; only 4.3 GFLOP)
// ---------------------------------------------------------------------------
__global__ __launch_bounds__(256) void rgemm_kernel(const float* Psi, const float* M, float* R) {
    __shared__ float sP[32][17];
    __shared__ float sM[32][17];
    const int tx = threadIdx.x, ty = threadIdx.y;
    const int a0 = blockIdx.y * 16, b0 = blockIdx.x * 16;
    float acc = 0.f;
    for (int k0 = 0; k0 < NN; k0 += 32) {
        #pragma unroll
        for (int s = 0; s < 2; ++s) {
            int kk = ty + s * 16;
            sP[kk][tx] = Psi[(long)(k0 + kk) * KK + a0 + tx] * T_SCALE;
            sM[kk][tx] = M[(long)(k0 + kk) * KK + b0 + tx];
        }
        __syncthreads();
        #pragma unroll
        for (int kk = 0; kk < 32; ++kk) acc += sP[kk][ty] * sM[kk][tx];
        __syncthreads();
    }
    R[(long)(a0 + ty) * KK + b0 + tx] = acc;
}

// out[0] = -trace(R); out[1] = alpha * sum_{a<b} R[a][b]^2
__global__ __launch_bounds__(256) void final_kernel(const float* R, float* out) {
    __shared__ float st[256], ss[256];
    const int t = threadIdx.x;
    float tr = 0.f, sq = 0.f;
    for (int idx = t; idx < KK * KK; idx += 256) {
        int a = idx >> 9, b = idx & (KK - 1);
        float v = R[idx];
        if (a == b) tr += v;
        else if (b > a) sq += v * v;
    }
    st[t] = tr; ss[t] = sq; __syncthreads();
    for (int o = 128; o > 0; o >>= 1) {
        if (t < o) { st[t] += st[t + o]; ss[t] += ss[t + o]; }
        __syncthreads();
    }
    if (t == 0) { out[0] = -st[0]; out[1] = ALPHA * ss[0]; }
}

// ---------------------------------------------------------------------------
// Launcher
// ---------------------------------------------------------------------------
extern "C" void kernel_launch(void* const* d_in, const int* in_sizes, int n_in,
                              void* d_out, int out_size, void* d_ws, size_t ws_size,
                              hipStream_t stream) {
    const float* FF  = (const float*)d_in[0];   // (8192, 384)
    const float* Psi = (const float*)d_in[1];   // (8192, 512)
    const float* im  = (const float*)d_in[2];   // (8, 3, 32, 32)
    float* out = (float*)d_out;                 // 2 floats

    char* ws = (char*)d_ws;
    uint16_t* FFn   = (uint16_t*)(ws + OFF_FFN);
    uint16_t* A     = (uint16_t*)(ws + OFF_A);
    float*    deg   = (float*)   (ws + OFF_DEG);
    float*    Dv    = (float*)   (ws + OFF_DV);
    uint16_t* DpsiT = (uint16_t*)(ws + OFF_DPT);
    float*    M     = (float*)   (ws + OFF_M);
    float*    R     = (float*)   (ws + OFF_R);
    uint8_t*  kn0   = (uint8_t*) (ws + OFF_K0);
    uint8_t*  kn1   = (uint8_t*) (ws + OFF_K1);
    uint8_t*  Ap    = (uint8_t*) (ws + OFF_AP);

    zero_f32_kernel<<<32, 256, 0, stream>>>(deg, NN);
    rownorm_kernel<<<NN, 128, 0, stream>>>(FF, FFn);

    knn_kernel<20><<<32, 256, 0, stream>>>(im, kn0, 2.0f);
    knn_kernel<10><<<32, 256, 0, stream>>>(im, kn1, 0.1f);
    mutual_kernel<<<(BSI * HW * HW) / 256, 256, 0, stream>>>(kn0, kn1, Ap);

    gemm_a_kernel<<<dim3(NN / 128, NN / 128), 256, 0, stream>>>(FFn, Ap, A, deg);

    dcalc_kernel<<<32, 256, 0, stream>>>(deg, Dv);
    dpsit_kernel<<<(KK * NN) / 256, 256, 0, stream>>>(Psi, Dv, DpsiT);

    gemm_m_kernel<<<dim3(KK / 128, NN / 128), 256, 0, stream>>>(A, DpsiT, Dv, M);

    rgemm_kernel<<<dim3(KK / 16, KK / 16), dim3(16, 16), 0, stream>>>(Psi, M, R);
    final_kernel<<<1, 256, 0, stream>>>(R, out);
}
